// MambaBlock_69595650064573
// MI455X (gfx1250) — matrix operand order repaired and checked by reading, not compiled, not run on hardware
//
#include <hip/hip_runtime.h>
#include <cstdint>

typedef unsigned short u16;
typedef unsigned int   u32;
typedef __attribute__((ext_vector_type(16))) __bf16 v16bf;
typedef __attribute__((ext_vector_type(8)))  float  v8f;

#define D_MODEL 1024
#define D_STATE 16
#define D_CONV  4
#define D_INNER 2048
#define DT_RANK 64
#define BATCH   2
#define SEQ     2048
#define NTOK    (BATCH*SEQ)

__device__ __forceinline__ u16 f2bf(float x){
  u32 u = __builtin_bit_cast(u32, x);
  u32 r = u + 0x7fffu + ((u >> 16) & 1u);   // round-to-nearest-even
  return (u16)(r >> 16);
}

union FragAB { u32 u[8]; uint4 q[2]; v16bf v; };
union FragC  { float f[8]; v8f v; };

// ---------------------------------------------------------------------------
// GEMM (32x64 wave tile): C[M,N] = A[M,K](bf16,lda) * W[N,K](bf16,ldb)^T
// Used for the skinny x_proj GEMM (N=96) where occupancy > reuse.
// ---------------------------------------------------------------------------
__global__ void gemm_bf16_wmma32(const u16* __restrict__ A, int lda,
                                 const u16* __restrict__ W, int ldb,
                                 float* __restrict__ C, int ldc,
                                 int M, int N, int K, int tilesN)
{
  const int lane  = threadIdx.x & 31;
  const int wave  = blockIdx.x * (blockDim.x >> 5) + (threadIdx.x >> 5);
  const int total = (M >> 5) * tilesN;
  if (wave >= total) return;
  const int tm   = (wave / tilesN) << 5;   // 32-row tile
  const int tn   = (wave % tilesN) << 6;   // 64-col tile
  const int ln16 = lane & 15;
  const int g    = (lane >> 4) & 1;

  FragC acc[2][4];
  #pragma unroll
  for (int i = 0; i < 2; i++)
    #pragma unroll
    for (int j = 0; j < 4; j++)
      #pragma unroll
      for (int e = 0; e < 8; e++) acc[i][j].f[e] = 0.0f;

  int brow[4];
  #pragma unroll
  for (int j = 0; j < 4; j++){
    int r = tn + j * 16 + ln16;
    brow[j] = (r < N) ? r : (N - 1);       // clamp for partial N tile
  }

  const u16* aBase0 = A + (size_t)(tm + ln16) * lda + g * 8;
  const u16* aBase1 = aBase0 + (size_t)16 * lda;

  for (int kb = 0; kb < K; kb += 32){
    FragAB a0, a1, b[4];
    const u16* p0 = aBase0 + kb;
    a0.q[0] = *reinterpret_cast<const uint4*>(p0);
    a0.q[1] = *reinterpret_cast<const uint4*>(p0 + 16);
    const u16* p1 = aBase1 + kb;
    a1.q[0] = *reinterpret_cast<const uint4*>(p1);
    a1.q[1] = *reinterpret_cast<const uint4*>(p1 + 16);
    #pragma unroll
    for (int j = 0; j < 4; j++){
      const u16* pb = W + (size_t)brow[j] * ldb + kb + g * 8;
      b[j].q[0] = *reinterpret_cast<const uint4*>(pb);
      b[j].q[1] = *reinterpret_cast<const uint4*>(pb + 16);
    }
    #pragma unroll
    for (int j = 0; j < 4; j++){
      acc[0][j].v = __builtin_amdgcn_wmma_f32_16x16x32_bf16(
          false, a0.v, false, b[j].v, (short)0, acc[0][j].v, false, false);
      acc[1][j].v = __builtin_amdgcn_wmma_f32_16x16x32_bf16(
          false, a1.v, false, b[j].v, (short)0, acc[1][j].v, false, false);
    }
  }

  #pragma unroll
  for (int i = 0; i < 2; i++)
    #pragma unroll
    for (int j = 0; j < 4; j++){
      int col = tn + j * 16 + ln16;
      if (col < N){
        #pragma unroll
        for (int v = 0; v < 8; v++){
          int row = tm + i * 16 + v + g * 8;
          C[(size_t)row * ldc + col] = acc[i][j].f[v];
        }
      }
    }
}

// ---------------------------------------------------------------------------
// GEMM (64x64 wave tile): higher arithmetic intensity (~33 FLOP/byte) for the
// big GEMMs. 4 A-frags held in registers, B-frags streamed; 16 WMMA per K=32.
// Requires M % 64 == 0 (true: M = NTOK = 4096).
// ---------------------------------------------------------------------------
__global__ void gemm_bf16_wmma64(const u16* __restrict__ A, int lda,
                                 const u16* __restrict__ W, int ldb,
                                 float* __restrict__ C, int ldc,
                                 int M, int N, int K, int tilesN)
{
  const int lane  = threadIdx.x & 31;
  const int wave  = blockIdx.x * (blockDim.x >> 5) + (threadIdx.x >> 5);
  const int total = (M >> 6) * tilesN;
  if (wave >= total) return;
  const int tm   = (wave / tilesN) << 6;   // 64-row tile
  const int tn   = (wave % tilesN) << 6;   // 64-col tile
  const int ln16 = lane & 15;
  const int g    = (lane >> 4) & 1;

  FragC acc[4][4];
  #pragma unroll
  for (int i = 0; i < 4; i++)
    #pragma unroll
    for (int j = 0; j < 4; j++)
      #pragma unroll
      for (int e = 0; e < 8; e++) acc[i][j].f[e] = 0.0f;

  int brow[4];
  #pragma unroll
  for (int j = 0; j < 4; j++){
    int r = tn + j * 16 + ln16;
    brow[j] = (r < N) ? r : (N - 1);
  }

  const u16* aBase = A + (size_t)(tm + ln16) * lda + g * 8;

  for (int kb = 0; kb < K; kb += 32){
    FragAB a[4];
    #pragma unroll
    for (int i = 0; i < 4; i++){
      const u16* p = aBase + (size_t)(i * 16) * lda + kb;
      a[i].q[0] = *reinterpret_cast<const uint4*>(p);
      a[i].q[1] = *reinterpret_cast<const uint4*>(p + 16);
    }
    #pragma unroll
    for (int j = 0; j < 4; j++){
      FragAB b;
      const u16* pb = W + (size_t)brow[j] * ldb + kb + g * 8;
      b.q[0] = *reinterpret_cast<const uint4*>(pb);
      b.q[1] = *reinterpret_cast<const uint4*>(pb + 16);
      #pragma unroll
      for (int i = 0; i < 4; i++){
        acc[i][j].v = __builtin_amdgcn_wmma_f32_16x16x32_bf16(
            false, a[i].v, false, b.v, (short)0, acc[i][j].v, false, false);
      }
    }
  }

  #pragma unroll
  for (int i = 0; i < 4; i++)
    #pragma unroll
    for (int j = 0; j < 4; j++){
      int col = tn + j * 16 + ln16;
      if (col < N){
        #pragma unroll
        for (int v = 0; v < 8; v++){
          int row = tm + i * 16 + v + g * 8;
          C[(size_t)row * ldc + col] = acc[i][j].f[v];
        }
      }
    }
}

// ---------------------------------------------------------------------------
// RMSNorm over D_MODEL=1024, one block (256 thr) per token; emits bf16.
// ---------------------------------------------------------------------------
__global__ void rmsnorm_bf16(const float* __restrict__ x, const float* __restrict__ w,
                             u16* __restrict__ xn)
{
  const int t = blockIdx.x;
  const float* xr = x + (size_t)t * D_MODEL;
  __shared__ float red[256];
  float vals[4]; float s = 0.f;
  #pragma unroll
  for (int i = 0; i < 4; i++){ vals[i] = xr[threadIdx.x + i * 256]; s += vals[i] * vals[i]; }
  red[threadIdx.x] = s;
  __syncthreads();
  for (int off = 128; off > 0; off >>= 1){
    if (threadIdx.x < off) red[threadIdx.x] += red[threadIdx.x + off];
    __syncthreads();
  }
  const float inv = rsqrtf(red[0] * (1.0f / D_MODEL) + 1e-5f);
  #pragma unroll
  for (int i = 0; i < 4; i++){
    int d = threadIdx.x + i * 256;
    xn[(size_t)t * D_MODEL + d] = f2bf(vals[i] * inv * w[d]);
  }
}

__global__ void cvt_f32_bf16(const float* __restrict__ s, u16* __restrict__ d, int n){
  int i = blockIdx.x * blockDim.x + threadIdx.x;
  if (i < n) d[i] = f2bf(s[i]);
}

// dt slice of dbl [NTOK,96] -> contiguous bf16 [NTOK,64]
__global__ void dt_slice_bf16(const float* __restrict__ dbl, u16* __restrict__ dt){
  int i = blockIdx.x * blockDim.x + threadIdx.x;
  if (i >= NTOK * DT_RANK) return;
  int r = i >> 6, c = i & 63;
  dt[i] = f2bf(dbl[(size_t)r * 96 + c]);
}

// depthwise causal conv (width 4) + bias + SiLU; xi = xz[:, 0:D_INNER]
__global__ void conv_silu(const float* __restrict__ xz, const float* __restrict__ cw,
                          const float* __restrict__ cb, float* __restrict__ xc,
                          u16* __restrict__ xcb)
{
  int i = blockIdx.x * blockDim.x + threadIdx.x;
  if (i >= NTOK * D_INNER) return;
  int t = i / D_INNER;
  int d = i - t * D_INNER;
  int tl = t & (SEQ - 1);          // position within batch (causal boundary)
  float acc = cb[d];
  #pragma unroll
  for (int j = 0; j < D_CONV; j++){
    int tj = tl - (D_CONV - 1) + j;
    if (tj >= 0)
      acc += xz[(size_t)(t - (D_CONV - 1) + j) * (2 * D_INNER) + d] * cw[d * D_CONV + j];
  }
  float v = acc / (1.0f + expf(-acc));   // SiLU
  xc[i]  = v;
  xcb[i] = f2bf(v);
}

// ---------------------------------------------------------------------------
// Selective scan: one wave = (batch b, 2 channels) x 16 states.
// lane: d = 2*dp + lane/16, n = lane%16. softplus fused; epilogue fused.
// ---------------------------------------------------------------------------
__global__ void selective_scan(const float* __restrict__ ddraw,  // [NTOK,D_INNER] pre-softplus
                               const float* __restrict__ dtb,    // [D_INNER]
                               const float* __restrict__ dbl,    // [NTOK,96] (B at 64, C at 80)
                               const float* __restrict__ xc,     // [NTOK,D_INNER]
                               const float* __restrict__ xz,     // [NTOK,2*D_INNER] (z at +D_INNER)
                               const float* __restrict__ A_log,  // [D_INNER,16]
                               const float* __restrict__ Dp,     // [D_INNER]
                               u16* __restrict__ ybf)            // [NTOK,D_INNER]
{
  const int lane = threadIdx.x & 31;
  const int wave = blockIdx.x * (blockDim.x >> 5) + (threadIdx.x >> 5);
  if (wave >= BATCH * (D_INNER / 2)) return;
  const int b  = wave / (D_INNER / 2);
  const int dp = wave - b * (D_INNER / 2);
  const int d  = dp * 2 + (lane >> 4);
  const int n  = lane & 15;

  const float a      = -expf(A_log[d * D_STATE + n]);
  const float bias   = dtb[d];
  const float dscale = Dp[d];
  float h = 0.f;
  const size_t tok0 = (size_t)b * SEQ;

  for (int t = 0; t < SEQ; t++){
    const size_t tok = tok0 + t;
    float dd    = ddraw[tok * D_INNER + d] + bias;
    float delta = (dd > 20.f) ? dd : log1pf(expf(dd));       // softplus
    float Bn    = dbl[tok * 96 + DT_RANK + n];
    float Cn    = dbl[tok * 96 + DT_RANK + D_STATE + n];
    float xv    = xc[tok * D_INNER + d];
    h = expf(delta * a) * h + delta * Bn * xv;
    float p = h * Cn;
    p += __shfl_xor(p, 1);
    p += __shfl_xor(p, 2);
    p += __shfl_xor(p, 4);
    p += __shfl_xor(p, 8);
    if (n == 0){
      float z = xz[tok * (2 * D_INNER) + D_INNER + d];
      float y = (p + xv * dscale) * (z / (1.f + expf(-z)));  // (+xc*D) * silu(z)
      ybf[tok * D_INNER + d] = f2bf(y);
    }
  }
}

__global__ void residual_add(const float* __restrict__ x, const float* __restrict__ y,
                             float* __restrict__ out, int n){
  int i = blockIdx.x * blockDim.x + threadIdx.x;
  if (i < n) out[i] = x[i] + y[i];
}

// ---------------------------------------------------------------------------
extern "C" void kernel_launch(void* const* d_in, const int* in_sizes, int n_in,
                              void* d_out, int out_size, void* d_ws, size_t ws_size,
                              hipStream_t stream)
{
  (void)in_sizes; (void)n_in; (void)out_size; (void)ws_size;
  const float* x     = (const float*)d_in[0];
  const float* normw = (const float*)d_in[1];
  const float* w_in  = (const float*)d_in[2];   // [4096,1024]
  const float* convw = (const float*)d_in[3];   // [2048,4]
  const float* convb = (const float*)d_in[4];
  const float* w_xp  = (const float*)d_in[5];   // [96,2048]
  const float* w_dtp = (const float*)d_in[6];   // [2048,64]
  const float* dtb   = (const float*)d_in[7];
  const float* A_log = (const float*)d_in[8];
  const float* Dp    = (const float*)d_in[9];
  const float* w_out = (const float*)d_in[10];  // [1024,2048]
  float* out = (float*)d_out;

  char* ws = (char*)d_ws; size_t off = 0;
  auto alloc = [&](size_t bytes)->char*{
    char* p = ws + off; off += (bytes + 255) & ~(size_t)255; return p;
  };
  u16*   w_in_bf  = (u16*)  alloc((size_t)2*D_INNER*D_MODEL*sizeof(u16));
  u16*   w_xp_bf  = (u16*)  alloc((size_t)(DT_RANK+2*D_STATE)*D_INNER*sizeof(u16));
  u16*   w_dtp_bf = (u16*)  alloc((size_t)D_INNER*DT_RANK*sizeof(u16));
  u16*   w_out_bf = (u16*)  alloc((size_t)D_MODEL*D_INNER*sizeof(u16));
  u16*   xn_bf    = (u16*)  alloc((size_t)NTOK*D_MODEL*sizeof(u16));
  float* xz       = (float*)alloc((size_t)NTOK*2*D_INNER*sizeof(float));
  float* xcf      = (float*)alloc((size_t)NTOK*D_INNER*sizeof(float));
  u16*   xc_bf    = (u16*)  alloc((size_t)NTOK*D_INNER*sizeof(u16));
  float* dbl      = (float*)alloc((size_t)NTOK*96*sizeof(float));
  u16*   dt_bf    = (u16*)  alloc((size_t)NTOK*DT_RANK*sizeof(u16));
  float* ddraw    = (float*)alloc((size_t)NTOK*D_INNER*sizeof(float));
  u16*   y_bf     = (u16*)  alloc((size_t)NTOK*D_INNER*sizeof(u16));
  float* yout     = (float*)alloc((size_t)NTOK*D_MODEL*sizeof(float));

  // 1) weight conversions f32 -> bf16
  {
    int n1 = 2*D_INNER*D_MODEL;
    cvt_f32_bf16<<<(n1+255)/256, 256, 0, stream>>>(w_in, w_in_bf, n1);
    int n2 = (DT_RANK+2*D_STATE)*D_INNER;
    cvt_f32_bf16<<<(n2+255)/256, 256, 0, stream>>>(w_xp, w_xp_bf, n2);
    int n3 = D_INNER*DT_RANK;
    cvt_f32_bf16<<<(n3+255)/256, 256, 0, stream>>>(w_dtp, w_dtp_bf, n3);
    int n4 = D_MODEL*D_INNER;
    cvt_f32_bf16<<<(n4+255)/256, 256, 0, stream>>>(w_out, w_out_bf, n4);
  }

  // 2) RMSNorm -> bf16
  rmsnorm_bf16<<<NTOK, 256, 0, stream>>>(x, normw, xn_bf);

  // 3) in_proj: xz = xn @ w_in^T   [4096 x 4096], K=1024  (64x64 tiles)
  {
    int N = 2*D_INNER, K = D_MODEL, tilesN = (N + 63) / 64;
    int waves = (NTOK/64) * tilesN;
    gemm_bf16_wmma64<<<(waves+3)/4, 128, 0, stream>>>(xn_bf, K, w_in_bf, K, xz, N,
                                                      NTOK, N, K, tilesN);
  }

  // 4) conv + SiLU
  {
    int n = NTOK * D_INNER;
    conv_silu<<<(n+255)/256, 256, 0, stream>>>(xz, convw, convb, xcf, xc_bf);
  }

  // 5) x_proj: dbl = xc @ w_xp^T   [4096 x 96], K=2048  (32x64 tiles, partial N)
  {
    int N = DT_RANK + 2*D_STATE, K = D_INNER, tilesN = (N + 63) / 64;
    int waves = (NTOK/32) * tilesN;
    gemm_bf16_wmma32<<<(waves+3)/4, 128, 0, stream>>>(xc_bf, K, w_xp_bf, K, dbl, N,
                                                      NTOK, N, K, tilesN);
  }

  // 6) dt slice -> bf16
  {
    int n = NTOK * DT_RANK;
    dt_slice_bf16<<<(n+255)/256, 256, 0, stream>>>(dbl, dt_bf);
  }

  // 7) dt_proj: ddraw = dt @ w_dtp^T  [4096 x 2048], K=64  (64x64 tiles)
  {
    int N = D_INNER, K = DT_RANK, tilesN = (N + 63) / 64;
    int waves = (NTOK/64) * tilesN;
    gemm_bf16_wmma64<<<(waves+3)/4, 128, 0, stream>>>(dt_bf, K, w_dtp_bf, K, ddraw, N,
                                                      NTOK, N, K, tilesN);
  }

  // 8) selective scan + fused epilogue -> y bf16
  {
    int waves = BATCH * (D_INNER / 2);      // 2048 waves
    selective_scan<<<(waves+3)/4, 128, 0, stream>>>(ddraw, dtb, dbl, xcf, xz,
                                                    A_log, Dp, y_bf);
  }

  // 9) out_proj: yout = y @ w_out^T  [4096 x 1024], K=2048  (64x64 tiles)
  {
    int N = D_MODEL, K = D_INNER, tilesN = (N + 63) / 64;
    int waves = (NTOK/64) * tilesN;
    gemm_bf16_wmma64<<<(waves+3)/4, 128, 0, stream>>>(y_bf, K, w_out_bf, K, yout, N,
                                                      NTOK, N, K, tilesN);
  }

  // 10) residual
  {
    int n = NTOK * D_MODEL;
    residual_add<<<(n+255)/256, 256, 0, stream>>>(x, yout, out, n);
  }
}
